// cost_volumn_82506321756235
// MI455X (gfx1250) — matrix operand order repaired and checked by reading, not compiled, not run on hardware
//
#include <hip/hip_runtime.h>

typedef __attribute__((ext_vector_type(16))) _Float16 v16h;
typedef __attribute__((ext_vector_type(8)))  float    v8f;
typedef __attribute__((ext_vector_type(4)))  float    v4f;

#define NB   8     // batch
#define HN   128   // height
#define WN   256   // width
#define CN   128   // channels
#define DR   4     // displacement radius
#define NOFF 9     // offsets per dim -> 81 total
#define WT   16    // w-tile handled by one wave
#define WAVES 4    // waves per block

// ---- A-matrix fragment loader (16x32 f16, ISA 7.12.2 "16-bit A-Matrix") ----
// Lane l holds row M = l%16. halves 0..7 = K c0+{0..7}, halves 8..15 = K
// c0+{16..23}, where c0 = kb*32 + (l>=16 ? 8 : 0). `p` points at channel c0.
__device__ __forceinline__ v16h load_fragA_f32(const float* __restrict__ p) {
  v16h f;
  v4f x0 = *(const v4f*)(p + 0);
  v4f x1 = *(const v4f*)(p + 4);
  v4f x2 = *(const v4f*)(p + 16);
  v4f x3 = *(const v4f*)(p + 20);
#pragma unroll
  for (int i = 0; i < 4; ++i) {
    f[i]      = (_Float16)x0[i];
    f[4 + i]  = (_Float16)x1[i];
    f[8 + i]  = (_Float16)x2[i];
    f[12 + i] = (_Float16)x3[i];
  }
  return f;
}

// ---- B-matrix fragment loader (32x16 f16, ISA 7.12.4 B-matrix layout) ----
// Lane l holds column N = l%16; halves 0..7 = K cb+{0..7}, halves 8..15 = K
// cb+{16..23}, where cb = kb*32 + (l>=16 ? 8 : 0). `p` points at channel cb.
__device__ __forceinline__ v16h load_fragB_f32(const float* __restrict__ p, bool valid) {
  v16h f;
  if (valid) {
    v4f x0 = *(const v4f*)(p + 0);
    v4f x1 = *(const v4f*)(p + 4);
    v4f x2 = *(const v4f*)(p + 16);
    v4f x3 = *(const v4f*)(p + 20);
#pragma unroll
    for (int i = 0; i < 4; ++i) {
      f[i]      = (_Float16)x0[i];
      f[4 + i]  = (_Float16)x1[i];
      f[8 + i]  = (_Float16)x2[i];
      f[12 + i] = (_Float16)x3[i];
    }
  } else {
#pragma unroll
    for (int i = 0; i < 16; ++i) f[i] = (_Float16)0.0f;
  }
  return f;
}

__global__ __launch_bounds__(WAVES * 32)
void costvol_wmma_kernel(const float* __restrict__ f1,
                         const float* __restrict__ f2,
                         float* __restrict__ out) {
  // Per-wave private LDS slab: 16x32 f32 C tile (padded stride 33).
  __shared__ float lds[WAVES][WT][33];

  const int tid  = (int)threadIdx.x;
  const int wave = tid >> 5;
  const int lane = tid & 31;

  // Tile id -> (n, h, w-tile). W/WT = 16, H = 128.
  const int t  = (int)blockIdx.x * WAVES + wave;
  const int wt = t & 15;
  const int h  = (t >> 4) & 127;
  const int n  = t >> 11;
  const int w0 = wt * WT;

  const int idx     = lane & 15;           // row (A) / column (B) index
  const int khalfA  = (lane >> 4) << 3;    // A: 0 or 8
  const int khalfB  = (lane >> 4) << 3;    // B: 0 or 8

  // ---- A fragments: f1[n, h, w0+idx, :], 4 K-blocks of 32 channels ----
  const float* a_base = f1 + (size_t)((n * HN + h) * WN + (w0 + idx)) * CN + khalfA;
  v16h a[4];
#pragma unroll
  for (int kb = 0; kb < 4; ++kb) a[kb] = load_fragA_f32(a_base + kb * 32);

  // B tile columns: tile0 covers w' = w0-4 .. w0+11, tile1 covers w0+12 .. w0+27
  const int  ww0 = w0 - DR + idx;
  const int  ww1 = w0 - DR + 16 + idx;
  const bool v0  = (ww0 >= 0) && (ww0 < WN);
  const bool v1  = (ww1 >= 0) && (ww1 < WN);

  const int rbase = (lane >> 4) << 3;      // C-tile row group for this lane half

  for (int di = 0; di < NOFF; ++di) {
    const int h2 = h + di - DR;            // f2 source row (wave-uniform)

    // Prefetch next di's f2 row while we compute this one.
    if (di < NOFF - 1) {
      const int h3 = h2 + 1;
      if (h3 >= 0 && h3 < HN && v0)
        __builtin_prefetch(f2 + (size_t)((n * HN + h3) * WN + ww0) * CN, 0, 0);
    }

    v8f c0 = {};
    v8f c1 = {};
    if (h2 >= 0 && h2 < HN) {              // uniform branch: EXEC all-ones at WMMA
      const float* b_row = f2 + (size_t)((n * HN + h2) * WN) * CN;
      const float* b0p   = b_row + (size_t)ww0 * CN + khalfB;
      const float* b1p   = b_row + (size_t)ww1 * CN + khalfB;
#pragma unroll
      for (int kb = 0; kb < 4; ++kb) {
        v16h b0 = load_fragB_f32(b0p + kb * 32, v0);
        c0 = __builtin_amdgcn_wmma_f32_16x16x32_f16(false, a[kb], false, b0,
                                                    (short)0, c0, false, false);
        v16h b1 = load_fragB_f32(b1p + kb * 32, v1);
        c1 = __builtin_amdgcn_wmma_f32_16x16x32_f16(false, a[kb], false, b1,
                                                    (short)0, c1, false, false);
      }
    }

    // ---- Spill the 16x32 C tile to LDS (C layout: vgpr r -> row rbase+r,
    // lane%16 -> column), then gather the 9-diagonal band. ----
#pragma unroll
    for (int r = 0; r < 8; ++r) {
      lds[wave][rbase + r][idx]      = c0[r];
      lds[wave][rbase + r][16 + idx] = c1[r];
    }
    asm volatile("s_wait_dscnt 0" ::: "memory");  // wave-local LDS RAW fence

    // Band element (w, dj) lives at C[w][w + dj]; 144 elements over 32 lanes.
#pragma unroll
    for (int e = 0; e < 5; ++e) {
      const int el = lane + e * 32;
      if (el < 144) {
        const int w  = el / 9;
        const int dj = el % 9;
        const float v = lds[wave][w][w + dj] * (1.0f / 128.0f);
        out[(size_t)((n * HN + h) * WN + (w0 + w)) * 81 + di * 9 + dj] = v;
      }
    }
    asm volatile("s_wait_dscnt 0" ::: "memory");  // WAR fence before next di
  }
}

extern "C" void kernel_launch(void* const* d_in, const int* in_sizes, int n_in,
                              void* d_out, int out_size, void* d_ws, size_t ws_size,
                              hipStream_t stream) {
  (void)in_sizes; (void)n_in; (void)out_size; (void)d_ws; (void)ws_size;
  const float* f1 = (const float*)d_in[0];
  const float* f2 = (const float*)d_in[1];
  float*       o  = (float*)d_out;

  const int tiles  = NB * HN * (WN / WT);   // 8*128*16 = 16384 wave-tiles
  const int blocks = tiles / WAVES;         // 4096 blocks of 4 waves
  costvol_wmma_kernel<<<blocks, WAVES * 32, 0, stream>>>(f1, f2, o);
}
